// MultiHeadAttention_386547056801
// MI455X (gfx1250) — matrix operand order repaired and checked by reading, not compiled, not run on hardware
//
#include <hip/hip_runtime.h>
#include <hip/hip_bf16.h>
#include <math.h>
#include <stdint.h>

// ---------------------------------------------------------------------------
// MHA forward for MI455X (gfx1250, wave32, WMMA + TDM).
// Roofline: ~142 GFLOP vs ~100 MB HBM traffic -> matrix-core bound; f16 WMMA
// (16x16x32, fp32 accum) everywhere. Flash attention keeps K/V per (b,h) in
// L2 (1 MB f16) and stages 32-key blocks into LDS with tensor_load_to_lds
// (TENSORcnt, double-buffered) so all 8 waves of a block share one DMA.
// ---------------------------------------------------------------------------

#define SLEN   4096
#define BSZ    2
#define NH     12
#define DM     768
#define DK     64
#define NKB    (SLEN / 32)

typedef _Float16 v8h  __attribute__((ext_vector_type(8)));
typedef _Float16 v16h __attribute__((ext_vector_type(16)));
typedef float    v8f  __attribute__((ext_vector_type(8)));
typedef unsigned int u32x4 __attribute__((ext_vector_type(4)));
typedef int          i32x8 __attribute__((ext_vector_type(8)));
typedef int          i32x4 __attribute__((ext_vector_type(4)));

// Workspace layout (units: _Float16 elements), total ~52.5 MB
#define WQT_OFF  ((size_t)0)
#define WMAT_SZ  ((size_t)NH * DK * DM)
#define WKT_OFF  (WQT_OFF + WMAT_SZ)
#define WVT_OFF  (WKT_OFF + WMAT_SZ)
#define WOT_OFF  (WVT_OFF + WMAT_SZ)
#define QH_OFF   (WOT_OFF + (size_t)DM * DM)
#define QKV_SZ   ((size_t)BSZ * NH * SLEN * DK)
#define KH_OFF   (QH_OFF + QKV_SZ)
#define VT_OFF   (KH_OFF + QKV_SZ)
#define AO_OFF   (VT_OFF + QKV_SZ)

#if __has_builtin(__builtin_amdgcn_tensor_load_to_lds)
#define MHA_TDM 1
#else
#define MHA_TDM 0
#endif

__device__ __forceinline__ v16h cat16(v8h lo, v8h hi) {
    return __builtin_shufflevector(lo, hi, 0,1,2,3,4,5,6,7,8,9,10,11,12,13,14,15);
}

// A fragment (16x32 f16, MxK), source row-major [row][k].
// Lane L: row = L&15; lanes 0-15 hold K {0..7,16..23}, lanes 16-31 {8..15,24..31}.
__device__ __forceinline__ v16h load_a_f16(const _Float16* base, int rowStride, int lane) {
    int m  = lane & 15;
    int k0 = (lane < 16) ? 0 : 8;
    const _Float16* p = base + (size_t)m * rowStride + k0;
    return cat16(*(const v8h*)(p), *(const v8h*)(p + 16));
}

// A fragment from fp32 source with convert-to-f16 on load.
__device__ __forceinline__ v16h load_a_f32cvt(const float* base, int rowStride, int lane) {
    int m  = lane & 15;
    int k0 = (lane < 16) ? 0 : 8;
    const float* p = base + (size_t)m * rowStride + k0;
    v8h lo, hi;
#pragma unroll
    for (int i = 0; i < 8; ++i) { lo[i] = (_Float16)p[i]; hi[i] = (_Float16)p[i + 16]; }
    return cat16(lo, hi);
}

// B fragment (32x16 f16, KxN) from source laid out src[n][k] (contiguous K per n).
// Lane L: n = L&15; lanes 0-15 hold K {0..7,16..23}, lanes 16-31 {8..15,24..31}.
__device__ __forceinline__ v16h load_b_f16(const _Float16* base, int colStride, int lane) {
    const _Float16* p = base + (size_t)(lane & 15) * colStride + ((lane >> 4) << 3);
    return cat16(*(const v8h*)(p), *(const v8h*)(p + 16));
}

__device__ __forceinline__ v8f wmma16(v16h a, v16h b, v8f c) {
    return __builtin_amdgcn_wmma_f32_16x16x32_f16(false, a, false, b, (short)0, c, false, false);
}

// Wave-private LDS ordering (LDS is in-order per wave; just stop the compiler
// from reordering around the cross-lane staging).
__device__ __forceinline__ void wave_lds_fence() {
    __builtin_amdgcn_fence(__ATOMIC_RELEASE, "wavefront");
    __builtin_amdgcn_wave_barrier();
    __builtin_amdgcn_fence(__ATOMIC_ACQUIRE, "wavefront");
}

#if MHA_TDM
// Issue one 2D TDM tile load: global (data_size=2B) -> LDS.
// D# per CDNA5 ISA 8.3/8.4: group0 {count=1, lds_addr, global_addr, type=2},
// group1 {data_size, tensor_dim0/1, tile_dim0/1, tensor_dim0_stride}.
// This toolchain's builtin is the 6-arg form:
//   (u32x4 g0, i32x8 g1, i32x4 g2, i32x4 g3, i32x8 extra, i32 cpol)
__device__ __forceinline__ void tdm_load_2d(uint32_t lds_off, const void* gptr,
                                            uint32_t tdim0, uint32_t tdim1,
                                            uint32_t til0, uint32_t til1,
                                            uint32_t stride0) {
    uint64_t ga = (uint64_t)(uintptr_t)gptr;
    u32x4 g0;
    g0[0] = 1u;                                              // count=1, user mode
    g0[1] = lds_off;                                         // lds_addr (bytes)
    g0[2] = (uint32_t)ga;                                    // global_addr[31:0]
    g0[3] = (uint32_t)((ga >> 32) & 0x1FFFFFFu) | (2u << 30);// addr[56:32] | type=2
    i32x8 g1;
    g1[0] = (int)(1u << 16);                                 // data_size = 2 bytes
    g1[1] = (int)((tdim0 & 0xFFFFu) << 16);                  // tensor_dim0[15:0]
    g1[2] = (int)((tdim0 >> 16) | ((tdim1 & 0xFFFFu) << 16));// dim0[31:16]|dim1[15:0]
    g1[3] = (int)((tdim1 >> 16) | (til0 << 16));             // dim1[31:16]|tile_dim0
    g1[4] = (int)(til1 & 0xFFFFu);                           // tile_dim1 (tile_dim2=0)
    g1[5] = (int)stride0;                                    // tensor_dim0_stride[31:0]
    g1[6] = 0;
    g1[7] = 0;
    i32x4 z4 = {0, 0, 0, 0};
    i32x8 z8 = {0, 0, 0, 0, 0, 0, 0, 0};
    __builtin_amdgcn_tensor_load_to_lds(g0, g1, z4, z4, z8, 0);
}
#endif

__device__ __forceinline__ uint32_t lds_byte_off(const void* p) {
    // Generic pointer to __shared__: addr[31:0] is the LDS byte offset.
    return (uint32_t)(uintptr_t)p;
}

// ---------------------------------------------------------------------------
// Kernel 1: weights f32 -> f16, transposed for contiguous B-fragment loads.
// ---------------------------------------------------------------------------
__global__ void mha_cvt_weights(const float* __restrict__ WQ, const float* __restrict__ WK,
                                const float* __restrict__ WV, const float* __restrict__ WO,
                                _Float16* __restrict__ wqt, _Float16* __restrict__ wkt,
                                _Float16* __restrict__ wvt, _Float16* __restrict__ wot) {
    int nt = gridDim.x * blockDim.x;
    int n1 = NH * DM * DK;
    for (int i = blockIdx.x * blockDim.x + threadIdx.x; i < n1; i += nt) {
        int h = i / (DM * DK);
        int r = (i / DK) % DM;
        int c = i % DK;
        size_t o = ((size_t)h * DK + c) * DM + r;
        wqt[o] = (_Float16)WQ[i];
        wkt[o] = (_Float16)WK[i];
        wvt[o] = (_Float16)WV[i];
    }
    int n2 = DM * DM;
    for (int i = blockIdx.x * blockDim.x + threadIdx.x; i < n2; i += nt) {
        int r = i / DM, c = i % DM;
        wot[(size_t)c * DM + r] = (_Float16)WO[i];
    }
}

// ---------------------------------------------------------------------------
// Kernel 2: per-head QKV projection; one wave = one 16x64 f16 output tile.
// q/k -> [b,h,s,dk]; v -> transposed [b,h,dk,s]. 18432 waves / 2304 blocks.
// ---------------------------------------------------------------------------
__global__ __launch_bounds__(256) void mha_proj_qkv(
        const float* __restrict__ Q, const float* __restrict__ K, const float* __restrict__ V,
        const _Float16* __restrict__ wqt, const _Float16* __restrict__ wkt,
        const _Float16* __restrict__ wvt,
        _Float16* __restrict__ qh, _Float16* __restrict__ kh, _Float16* __restrict__ vt) {
    int lane = threadIdx.x & 31;
    int gw   = blockIdx.x * 8 + (threadIdx.x >> 5);
    int mat  = gw / (BSZ * NH * (SLEN / 16));
    int rem  = gw % (BSZ * NH * (SLEN / 16));
    int t    = rem & 255;
    int bh   = rem >> 8;
    int h    = bh % NH;
    int b    = bh / NH;

    const float*    src = (mat == 0 ? Q : (mat == 1 ? K : V)) + ((size_t)b * SLEN + t * 16) * DM;
    const _Float16* wt  = (mat == 0 ? wqt : (mat == 1 ? wkt : wvt)) + (size_t)h * DK * DM;

    v8f c[4] = {};
#pragma unroll 4
    for (int ks = 0; ks < DM / 32; ++ks) {
        v16h a  = load_a_f32cvt(src + ks * 32, DM, lane);
        v16h b0 = load_b_f16(wt + (size_t)0 * 16 * DM + ks * 32, DM, lane);
        v16h b1 = load_b_f16(wt + (size_t)1 * 16 * DM + ks * 32, DM, lane);
        v16h b2 = load_b_f16(wt + (size_t)2 * 16 * DM + ks * 32, DM, lane);
        v16h b3 = load_b_f16(wt + (size_t)3 * 16 * DM + ks * 32, DM, lane);
        c[0] = wmma16(a, b0, c[0]);
        c[1] = wmma16(a, b1, c[1]);
        c[2] = wmma16(a, b2, c[2]);
        c[3] = wmma16(a, b3, c[3]);
    }

    int mhi = (lane >> 4) << 3;
    int n   = lane & 15;
    if (mat < 2) {
        _Float16* dst = (mat == 0 ? qh : kh) + ((size_t)bh * SLEN + t * 16) * DK;
#pragma unroll
        for (int ns = 0; ns < 4; ++ns)
#pragma unroll
            for (int r = 0; r < 8; ++r)
                dst[(size_t)(r + mhi) * DK + ns * 16 + n] = (_Float16)c[ns][r];
    } else {
        _Float16* dst = vt + (size_t)bh * DK * SLEN + t * 16;
#pragma unroll
        for (int ns = 0; ns < 4; ++ns)
#pragma unroll
            for (int r = 0; r < 8; ++r)
                dst[(size_t)(ns * 16 + n) * SLEN + r + mhi] = (_Float16)c[ns][r];
    }
}

// ---------------------------------------------------------------------------
// Kernel 3: flash attention. Block = 8 waves, one (b,h), 128 query rows.
// K/V 32-key blocks staged into LDS by TDM (wave 0), double-buffered so DMA
// of block kb+2 overlaps WMMA of block kb+1; consumers gate on s_wait_tensorcnt.
// Online softmax stats replicated across 16-lane half-groups (C-frag rows).
// ---------------------------------------------------------------------------
__global__ __launch_bounds__(256) void mha_flash(
        const _Float16* __restrict__ qh, const _Float16* __restrict__ kh,
        const _Float16* __restrict__ vt, _Float16* __restrict__ attnout) {
    __shared__ __align__(16) _Float16 ldsK[2][32 * DK];   // [key][d]
    __shared__ __align__(16) _Float16 ldsV[2][DK * 32];   // [d][key]
    __shared__ __align__(16) _Float16 ldsP[8][16 * 32];   // per-wave P tile

    int lane = threadIdx.x & 31;
    int wave = threadIdx.x >> 5;
    int gw   = blockIdx.x * 8 + wave;
    int qt   = gw & 255;
    int bh   = gw >> 8;
    int h    = bh % NH;
    int b    = bh / NH;

    const _Float16* qbase = qh + ((size_t)bh * SLEN + qt * 16) * DK;
    const _Float16* kbase = kh + (size_t)bh * SLEN * DK;
    const _Float16* vbase = vt + (size_t)bh * DK * SLEN;

    v16h aq0 = load_a_f16(qbase,      DK, lane);
    v16h aq1 = load_a_f16(qbase + 32, DK, lane);

    float m[8], l[8];
    v8f   o[4] = {};
#pragma unroll
    for (int r = 0; r < 8; ++r) { m[r] = -1.0e30f; l[r] = 0.0f; }

    const float sc = 0.125f;       // 1/sqrt(DK)
    int mhi = (lane >> 4) << 3;
    int n   = lane & 15;

#if MHA_TDM
    uint32_t kOff[2] = { lds_byte_off(&ldsK[0][0]), lds_byte_off(&ldsK[1][0]) };
    uint32_t vOff[2] = { lds_byte_off(&ldsV[0][0]), lds_byte_off(&ldsV[1][0]) };
    if (wave == 0) {
        // Prefetch key blocks 0 and 1.
#pragma unroll
        for (int p = 0; p < 2; ++p) {
            tdm_load_2d(kOff[p], kbase + (size_t)p * 32 * DK, DK, SLEN, DK, 32, DK);
            tdm_load_2d(vOff[p], vbase + (size_t)p * 32,      SLEN, DK, 32, DK, SLEN);
        }
    }
#endif

    for (int kb = 0; kb < NKB; ++kb) {
        int buf = kb & 1;
#if MHA_TDM
        if (wave == 0) {
            if (kb + 1 < NKB) __builtin_amdgcn_s_wait_tensorcnt(2);  // next block may be in flight
            else              __builtin_amdgcn_s_wait_tensorcnt(0);
        }
#else
        {   // cooperative synchronous stage (fallback, no TDM builtin)
            const uint32_t* gk = (const uint32_t*)(kbase + (size_t)kb * 32 * DK);
            uint32_t* lk = (uint32_t*)&ldsK[buf][0];
            for (int i = threadIdx.x; i < 32 * DK / 2; i += 256) lk[i] = gk[i];
            uint32_t* lv = (uint32_t*)&ldsV[buf][0];
            const _Float16* vg = vbase + (size_t)kb * 32;
            for (int i = threadIdx.x; i < DK * 16; i += 256) {
                int row = i >> 4, col = i & 15;
                lv[row * 16 + col] = *(const uint32_t*)(vg + (size_t)row * SLEN + col * 2);
            }
        }
#endif
        __syncthreads();   // K/V tile for `buf` resident in LDS

        // scores 16x32: c0 = keys 0..15, c1 = keys 16..31 of the block
        const _Float16* kt = &ldsK[buf][0];
        v16h bk00 = load_b_f16(kt,                DK, lane);
        v16h bk10 = load_b_f16(kt + 32,           DK, lane);
        v16h bk01 = load_b_f16(kt + 16 * DK,      DK, lane);
        v16h bk11 = load_b_f16(kt + 16 * DK + 32, DK, lane);
        v8f c0 = {}, c1 = {};
        c0 = wmma16(aq0, bk00, c0);
        c0 = wmma16(aq1, bk10, c0);
        c1 = wmma16(aq0, bk01, c1);
        c1 = wmma16(aq1, bk11, c1);

        float alpha[8];
#pragma unroll
        for (int r = 0; r < 8; ++r) {
            float s0 = c0[r] * sc, s1 = c1[r] * sc;
            float mx = fmaxf(s0, s1);
#pragma unroll
            for (int off = 8; off; off >>= 1) mx = fmaxf(mx, __shfl_xor(mx, off, 32));
            float mn = fmaxf(m[r], mx);
            float a  = __expf(m[r] - mn);
            float p0 = __expf(s0 - mn);
            float p1 = __expf(s1 - mn);
            float rs = p0 + p1;
#pragma unroll
            for (int off = 8; off; off >>= 1) rs += __shfl_xor(rs, off, 32);
            l[r] = l[r] * a + rs;
            m[r] = mn;
            alpha[r] = a;
            int mrow = r + mhi;
            ldsP[wave][mrow * 32 + n]      = (_Float16)p0;
            ldsP[wave][mrow * 32 + 16 + n] = (_Float16)p1;
        }
        wave_lds_fence();   // P staging is wave-private; LDS is in-order per wave

        v16h ap;
        {
            int mr = lane & 15;
            int k0 = (lane < 16) ? 0 : 8;
            const _Float16* p = &ldsP[wave][mr * 32 + k0];
            ap = cat16(*(const v8h*)p, *(const v8h*)(p + 16));
        }

        // O = diag(alpha)*O + P @ V_block
        const _Float16* vtile = &ldsV[buf][0];
        v16h bv0 = load_b_f16(vtile + 0 * 16 * 32, 32, lane);
        v16h bv1 = load_b_f16(vtile + 1 * 16 * 32, 32, lane);
        v16h bv2 = load_b_f16(vtile + 2 * 16 * 32, 32, lane);
        v16h bv3 = load_b_f16(vtile + 3 * 16 * 32, 32, lane);
#pragma unroll
        for (int ns = 0; ns < 4; ++ns) {
            v8f t = o[ns];
#pragma unroll
            for (int r = 0; r < 8; ++r) t[r] *= alpha[r];
            o[ns] = wmma16(ap, (ns == 0 ? bv0 : ns == 1 ? bv1 : ns == 2 ? bv2 : bv3), t);
        }
        __syncthreads();   // all waves done with `buf`; safe to refill

#if MHA_TDM
        if (wave == 0 && kb + 2 < NKB) {
            tdm_load_2d(kOff[buf], kbase + (size_t)(kb + 2) * 32 * DK, DK, SLEN, DK, 32, DK);
            tdm_load_2d(vOff[buf], vbase + (size_t)(kb + 2) * 32,      SLEN, DK, 32, DK, SLEN);
        }
#endif
    }

    // Normalize and write concatenated-head f16 output [b, s, h*64+d].
    int row0 = qt * 16;
#pragma unroll
    for (int r = 0; r < 8; ++r) {
        float inv = 1.0f / l[r];
        int row = row0 + r + mhi;
#pragma unroll
        for (int ns = 0; ns < 4; ++ns)
            attnout[((size_t)b * SLEN + row) * DM + h * DK + ns * 16 + n] =
                (_Float16)(o[ns][r] * inv);
    }
}

// ---------------------------------------------------------------------------
// Kernel 4: output projection [B*S,768] f16 x WO^T f16 -> fp32 out.
// ---------------------------------------------------------------------------
__global__ __launch_bounds__(256) void mha_oproj(
        const _Float16* __restrict__ x, const _Float16* __restrict__ wot,
        float* __restrict__ out) {
    int lane = threadIdx.x & 31;
    int gw   = blockIdx.x * 8 + (threadIdx.x >> 5);
    int ct   = gw % (DM / 64);
    int rt   = gw / (DM / 64);
    const _Float16* xb = x   + (size_t)rt * 16 * DM;
    const _Float16* wb = wot + (size_t)ct * 64 * DM;

    v8f c[4] = {};
#pragma unroll 4
    for (int ks = 0; ks < DM / 32; ++ks) {
        v16h a  = load_a_f16(xb + ks * 32, DM, lane);
        v16h b0 = load_b_f16(wb + (size_t)0 * 16 * DM + ks * 32, DM, lane);
        v16h b1 = load_b_f16(wb + (size_t)1 * 16 * DM + ks * 32, DM, lane);
        v16h b2 = load_b_f16(wb + (size_t)2 * 16 * DM + ks * 32, DM, lane);
        v16h b3 = load_b_f16(wb + (size_t)3 * 16 * DM + ks * 32, DM, lane);
        c[0] = wmma16(a, b0, c[0]);
        c[1] = wmma16(a, b1, c[1]);
        c[2] = wmma16(a, b2, c[2]);
        c[3] = wmma16(a, b3, c[3]);
    }
    int mhi = (lane >> 4) << 3;
    int n   = lane & 15;
#pragma unroll
    for (int ns = 0; ns < 4; ++ns)
#pragma unroll
        for (int r = 0; r < 8; ++r)
            out[((size_t)rt * 16 + r + mhi) * DM + ct * 64 + ns * 16 + n] = c[ns][r];
}

// ---------------------------------------------------------------------------
extern "C" void kernel_launch(void* const* d_in, const int* in_sizes, int n_in,
                              void* d_out, int out_size, void* d_ws, size_t ws_size,
                              hipStream_t stream) {
    const float* Q  = (const float*)d_in[0];
    const float* K  = (const float*)d_in[1];
    const float* V  = (const float*)d_in[2];
    const float* WQ = (const float*)d_in[3];
    const float* WK = (const float*)d_in[4];
    const float* WV = (const float*)d_in[5];
    const float* WO = (const float*)d_in[6];
    float* out = (float*)d_out;

    _Float16* ws  = (_Float16*)d_ws;      // needs ~52.5 MB of workspace
    _Float16* wqt = ws + WQT_OFF;
    _Float16* wkt = ws + WKT_OFF;
    _Float16* wvt = ws + WVT_OFF;
    _Float16* wot = ws + WOT_OFF;
    _Float16* qhp = ws + QH_OFF;
    _Float16* khp = ws + KH_OFF;
    _Float16* vtp = ws + VT_OFF;
    _Float16* aop = ws + AO_OFF;

    mha_cvt_weights<<<256, 256, 0, stream>>>(WQ, WK, WV, WO, wqt, wkt, wvt, wot);
    mha_proj_qkv<<<2304, 256, 0, stream>>>(Q, K, V, wqt, wkt, wvt, qhp, khp, vtp);
    mha_flash<<<768, 256, 0, stream>>>(qhp, khp, vtp, aop);
    mha_oproj<<<768, 256, 0, stream>>>(aop, wot, out);
}
